// FastWeightMemory_69690139345093
// MI455X (gfx1250) — matrix operand order repaired and checked by reading, not compiled, not run on hardware
//
#include <hip/hip_runtime.h>
#include <hip/hip_bf16.h>

// ---------------------------------------------------------------------------
// FastWeightMemory on MI455X (gfx1250, wave32, WMMA).
//   1) convert x, Wq, Wk, Wv, Wo  f32 -> bf16
//   2) qkv projection GEMMs, A panel staged in LDS (shared by 8 waves/block),
//      v_wmma_f32_16x16x32_bf16, templated epilogue (q->bf16, k/v->f32)
//   3) L2-normalize k,v rows; write TRANSPOSED bf16 (feature-major) so the
//      scan's outer-product WMMA fragments are contiguous per lane
//   4) persistent scan: 16 workgroups each own 16 rows of the 256x256 fp32
//      state in LDS; per chunk r = q @ M^T and M = decay*M + scale*v^T k
//   5) output projection GEMM (r @ Wo^T), A panel in LDS, f32 to d_out
// ---------------------------------------------------------------------------

typedef __bf16 bf16;
typedef bf16  bf16x8  __attribute__((ext_vector_type(8)));
typedef bf16  bf16x16 __attribute__((ext_vector_type(16)));
typedef float f32x4   __attribute__((ext_vector_type(4)));
typedef float f32x8   __attribute__((ext_vector_type(8)));

#define B_   4
#define S_   4096
#define H_   1024
#define M_   256
#define C_   64        // chunk size
#define NC_  64        // number of chunks
#define NTOK (B_ * S_) // 16384

// ---- WMMA fragment loaders (contiguous k = 16*(lane>>4) + s in A and B, wave32) ----------

// A matrix 16x32 bf16, source row-major [row][k] with leading dim ld.
// lane<16: row=lane,   K = {0..15}; lane>=16: row=lane-16, K = {16..31}
__device__ __forceinline__ bf16x16 load_frag_a(const bf16* base, int ld, int lane) {
  const int row  = lane & 15;
  const int ksel = (lane >> 4) * 16; // 0 or 16
  union { bf16x16 v; bf16x8 h[2]; } u;
  u.h[0] = *reinterpret_cast<const bf16x8*>(base + row * ld + ksel);
  u.h[1] = *reinterpret_cast<const bf16x8*>(base + row * ld + 8 + ksel);
  return u.v;
}

// B matrix 32x16 bf16 (KxN), source stored N-major [n][k] with leading dim ld.
// lane: col N = lane&15, 16 contiguous K at offset 16*(lane>=16).
__device__ __forceinline__ bf16x16 load_frag_b(const bf16* base, int ld, int lane) {
  const int n    = lane & 15;
  const int koff = (lane >> 4) * 16; // 0 or 16
  return *reinterpret_cast<const bf16x16*>(base + n * ld + koff);
}

__device__ __forceinline__ f32x8 wmma_bf16(bf16x16 a, bf16x16 b, f32x8 c) {
  return __builtin_amdgcn_wmma_f32_16x16x32_bf16(
      /*neg_a=*/false, a, /*neg_b=*/false, b,
      /*c_mod=*/(short)0, c, /*reuse_a=*/false, /*reuse_b=*/false);
}

// ---- 1) f32 -> bf16 conversion (4-wide) ------------------------------------
__global__ void cvt_f32_bf16(const float* __restrict__ in, bf16* __restrict__ out, int n) {
  int i = 4 * (blockIdx.x * blockDim.x + threadIdx.x);
  if (i + 3 < n) {
    f32x4 v = *reinterpret_cast<const f32x4*>(in + i);
    typedef bf16 bf16x4 __attribute__((ext_vector_type(4)));
    bf16x4 o; o[0] = (bf16)v[0]; o[1] = (bf16)v[1]; o[2] = (bf16)v[2]; o[3] = (bf16)v[3];
    *reinterpret_cast<bf16x4*>(out + i) = o;
  }
}

// ---- 2) projection GEMM: out[tok][n] = sum_h x[tok][h] * W[n][h] -----------
// grid(1024, 2), block 256 (8 waves). A panel (16 tokens x 1024) staged in LDS
// once per block; row pad 8 elems => row stride 2064 B = 516 dwords (== 4 mod 64,
// 16 conflict-free banks for the 16 A-fragment rows; 16-byte aligned).
#define XP_LD (H_ + 8)
template <bool WRITE_BF>
__global__ void proj_gemm(const bf16* __restrict__ x, const bf16* __restrict__ w,
                          float* __restrict__ out_f32, bf16* __restrict__ out_bf) {
  __shared__ bf16 Apan[16 * XP_LD];
  const int tid  = threadIdx.x;
  const int lane = tid & 31;
  const int wave = tid >> 5;
  const int n0 = (blockIdx.y * 8 + wave) * 16;
  const int tok0 = blockIdx.x * 16;

  // cooperative stage of the A panel: 16*1024 bf16 = 2048 16-byte vectors
  {
    const bf16* src = x + (size_t)tok0 * H_;
#pragma unroll
    for (int it = 0; it < 8; ++it) {
      const int i = tid + 256 * it;          // vector index
      const int row = i >> 7;                // / (H_/8)
      const int c8 = i & 127;
      *reinterpret_cast<bf16x8*>(&Apan[row * XP_LD + c8 * 8]) =
          *reinterpret_cast<const bf16x8*>(src + row * H_ + c8 * 8);
    }
  }
  __syncthreads();

  f32x8 acc{};
  const bf16* wb = w + (size_t)n0 * H_;
#pragma unroll 4
  for (int k0 = 0; k0 < H_; k0 += 32) {
    __builtin_prefetch(wb + k0 + 256, 0, 0);          // global_prefetch_b8
    bf16x16 a = load_frag_a(&Apan[k0], XP_LD, lane);  // ds_load_b128
    bf16x16 b = load_frag_b(wb + k0, H_, lane);
    acc = wmma_bf16(a, b, acc);
  }
  const int col = n0 + (lane & 15);
  const int rowoff = tok0 + 8 * (lane >> 4);
#pragma unroll
  for (int j = 0; j < 8; ++j) {
    const size_t idx = (size_t)(rowoff + j) * M_ + col;
    if constexpr (WRITE_BF) out_bf[idx] = (bf16)acc[j];
    else                    out_f32[idx] = acc[j];
  }
}

// ---- 3) L2-normalize rows of k,v; write transposed bf16 [feat][token] ------
__global__ void norm_transpose(const float* __restrict__ kf, const float* __restrict__ vf,
                               bf16* __restrict__ kT, bf16* __restrict__ vT) {
  const int lane = threadIdx.x & 31;
  const int wave = threadIdx.x >> 5;
  const int token = blockIdx.x * 8 + wave;
  const int m0 = lane * 8;

  const float* krow = kf + (size_t)token * M_ + m0;
  const float* vrow = vf + (size_t)token * M_ + m0;
  float kv[8], vv[8];
  float ks = 0.f, vs = 0.f;
#pragma unroll
  for (int j = 0; j < 8; ++j) {
    kv[j] = krow[j]; vv[j] = vrow[j];
    ks += kv[j] * kv[j]; vs += vv[j] * vv[j];
  }
#pragma unroll
  for (int off = 16; off > 0; off >>= 1) {
    ks += __shfl_xor(ks, off, 32);
    vs += __shfl_xor(vs, off, 32);
  }
  const float kinv = 1.0f / fmaxf(sqrtf(ks), 1e-12f);
  const float vinv = 1.0f / fmaxf(sqrtf(vs), 1e-12f);
#pragma unroll
  for (int j = 0; j < 8; ++j) {
    kT[(size_t)(m0 + j) * NTOK + token] = (bf16)(kv[j] * kinv);
    vT[(size_t)(m0 + j) * NTOK + token] = (bf16)(vv[j] * vinv);
  }
}

// ---- 4) chunked scan -------------------------------------------------------
// grid 16 workgroups; wg g owns state rows n in [16g, 16g+16).
// Mb row pad => 528 B = 132 dwords (== 4 mod 64, conflict-free; 16 B aligned).
#define MB_LD 264
__global__ void scan_kernel(const bf16* __restrict__ q, const bf16* __restrict__ kT,
                            const bf16* __restrict__ vT, const float* __restrict__ M0,
                            bf16* __restrict__ r_bf, float decay_c, float scale) {
  __shared__ float Ms[16][M_];
  __shared__ bf16  Mb[16][MB_LD];
  const int tid  = threadIdx.x;
  const int lane = tid & 31;
  const int wave = tid >> 5;
  const int nslice = blockIdx.x * 16;

  for (int e = tid; e < 16 * M_; e += 256)
    Ms[e >> 8][e & 255] = M0[(size_t)(nslice + (e >> 8)) * M_ + (e & 255)];
  __syncthreads();

  for (int t = 0; t < NC_; ++t) {
    // bf16 snapshot of the state for the r GEMM's B operand
    for (int e = tid; e < 16 * M_; e += 256)
      Mb[e >> 8][e & 255] = (bf16)Ms[e >> 8][e & 255];
    __syncthreads();

    // r[rows, nslice..+16) = q_chunk(256 x 256) @ Ms^T   (A = q, B from Mb)
    for (int tt = wave; tt < 16; tt += 8) {
      const int rr0 = tt * 16;
      const bf16* qbase = q + (size_t)((rr0 >> 6) * S_ + t * C_ + (rr0 & 63)) * M_;
      f32x8 acc{};
#pragma unroll
      for (int m0 = 0; m0 < M_; m0 += 32) {
        bf16x16 a  = load_frag_a(qbase + m0, M_, lane);
        bf16x16 bm = load_frag_b(&Mb[0][0] + m0, MB_LD, lane);
        acc = wmma_bf16(a, bm, acc);
      }
      const int col = nslice + (lane & 15);
      const int rowoff = 8 * (lane >> 4);
#pragma unroll
      for (int j = 0; j < 8; ++j) {
        const int rloc = rr0 + rowoff + j;
        const int token = (rloc >> 6) * S_ + t * C_ + (rloc & 63);
        r_bf[(size_t)token * M_ + col] = (bf16)acc[j];
      }
    }

    // M[n, m] = decay*M + scale * sum_rows v[row,n] * k[row,m]
    for (int mt = wave; mt < 16; mt += 8) {
      const int m0g = mt * 16;
      f32x8 acc{};
#pragma unroll
      for (int r0 = 0; r0 < 256; r0 += 32) {
        const int tokbase = (r0 >> 6) * S_ + t * C_ + (r0 & 63);
        bf16x16 a  = load_frag_a(vT + (size_t)nslice * NTOK + tokbase, NTOK, lane);
        bf16x16 bk = load_frag_b(kT + (size_t)m0g * NTOK + tokbase, NTOK, lane);
        acc = wmma_bf16(a, bk, acc);
      }
      const int mcol = m0g + (lane & 15);
      const int rowoff = 8 * (lane >> 4);
#pragma unroll
      for (int j = 0; j < 8; ++j) {
        const int nl = rowoff + j;
        Ms[nl][mcol] = decay_c * Ms[nl][mcol] + scale * acc[j];
      }
    }
    __syncthreads();
  }
}

// ---- 5) output GEMM: out[tok][h] = sum_m r[tok][m] * Wo[h][m] --------------
// grid(1024, 8), block 256; A panel (16 tokens x 256) staged in LDS.
// Row pad 8 => 528 B stride (conflict-free, 16 B aligned).
#define RP_LD (M_ + 8)
__global__ void out_gemm(const bf16* __restrict__ r, const bf16* __restrict__ wo,
                         float* __restrict__ out) {
  __shared__ bf16 Apan[16 * RP_LD];
  const int tid  = threadIdx.x;
  const int lane = tid & 31;
  const int wave = tid >> 5;
  const int h0 = (blockIdx.y * 8 + wave) * 16;
  const int tok0 = blockIdx.x * 16;

  {
    const bf16* src = r + (size_t)tok0 * M_;
#pragma unroll
    for (int it = 0; it < 2; ++it) {
      const int i = tid + 256 * it;          // 512 16-byte vectors
      const int row = i >> 5;                // / (M_/8)
      const int c8 = i & 31;
      *reinterpret_cast<bf16x8*>(&Apan[row * RP_LD + c8 * 8]) =
          *reinterpret_cast<const bf16x8*>(src + row * M_ + c8 * 8);
    }
  }
  __syncthreads();

  f32x8 acc{};
  const bf16* wb = wo + (size_t)h0 * M_;
#pragma unroll
  for (int m0 = 0; m0 < M_; m0 += 32) {
    bf16x16 a = load_frag_a(&Apan[m0], RP_LD, lane);
    bf16x16 b = load_frag_b(wb + m0, M_, lane);
    acc = wmma_bf16(a, b, acc);
  }
  const int col = h0 + (lane & 15);
  const int rowoff = tok0 + 8 * (lane >> 4);
#pragma unroll
  for (int j = 0; j < 8; ++j)
    out[(size_t)(rowoff + j) * H_ + col] = acc[j];
}

// ---------------------------------------------------------------------------
extern "C" void kernel_launch(void* const* d_in, const int* in_sizes, int n_in,
                              void* d_out, int out_size, void* d_ws, size_t ws_size,
                              hipStream_t stream) {
  const float* x  = (const float*)d_in[0];
  const float* Wq = (const float*)d_in[1];
  const float* Wk = (const float*)d_in[2];
  const float* Wv = (const float*)d_in[3];
  const float* Wo = (const float*)d_in[4];
  const float* M0 = (const float*)d_in[5];
  float* out = (float*)d_out;

  char* ws = (char*)d_ws;
  size_t off = 0;
  auto alloc = [&](size_t bytes) { char* p = ws + off; off += (bytes + 255) & ~size_t(255); return p; };
  bf16*  x_bf  = (bf16*)alloc((size_t)NTOK * H_ * 2);   // 32 MB
  bf16*  wq_bf = (bf16*)alloc((size_t)M_ * H_ * 2);
  bf16*  wk_bf = (bf16*)alloc((size_t)M_ * H_ * 2);
  bf16*  wv_bf = (bf16*)alloc((size_t)M_ * H_ * 2);
  bf16*  wo_bf = (bf16*)alloc((size_t)H_ * M_ * 2);
  bf16*  q_bf  = (bf16*)alloc((size_t)NTOK * M_ * 2);   // 8 MB
  float* k_f32 = (float*)alloc((size_t)NTOK * M_ * 4);  // 16 MB
  float* v_f32 = (float*)alloc((size_t)NTOK * M_ * 4);  // 16 MB
  bf16*  kT_bf = (bf16*)alloc((size_t)M_ * NTOK * 2);   // 8 MB
  bf16*  vT_bf = (bf16*)alloc((size_t)M_ * NTOK * 2);   // 8 MB
  bf16*  r_bf  = (bf16*)alloc((size_t)NTOK * M_ * 2);   // 8 MB
  (void)ws_size; (void)in_sizes; (void)n_in; (void)out_size;

  double d = 1.0;
  for (int i = 0; i < C_; ++i) d *= 0.99;
  const float decay_c = (float)d;
  const float scale = 1.0f / (B_ * C_);

  // 1) conversions
  {
    int n = NTOK * H_;
    cvt_f32_bf16<<<n / (256 * 4), 256, 0, stream>>>(x, x_bf, n);
    int nw = M_ * H_;
    cvt_f32_bf16<<<nw / (256 * 4), 256, 0, stream>>>(Wq, wq_bf, nw);
    cvt_f32_bf16<<<nw / (256 * 4), 256, 0, stream>>>(Wk, wk_bf, nw);
    cvt_f32_bf16<<<nw / (256 * 4), 256, 0, stream>>>(Wv, wv_bf, nw);
    cvt_f32_bf16<<<nw / (256 * 4), 256, 0, stream>>>(Wo, wo_bf, nw);
  }

  // 2) projections (WMMA, LDS-staged A panel)
  {
    dim3 grid(NTOK / 16, 2);
    proj_gemm<true ><<<grid, 256, 0, stream>>>(x_bf, wq_bf, nullptr, q_bf);
    proj_gemm<false><<<grid, 256, 0, stream>>>(x_bf, wk_bf, k_f32, nullptr);
    proj_gemm<false><<<grid, 256, 0, stream>>>(x_bf, wv_bf, v_f32, nullptr);
  }

  // 3) normalize + transpose
  norm_transpose<<<NTOK / 8, 256, 0, stream>>>(k_f32, v_f32, kT_bf, vT_bf);

  // 4) chunked scan (persistent, state in LDS, WMMA)
  scan_kernel<<<16, 256, 0, stream>>>(q_bf, kT_bf, vT_bf, M0, r_bf, decay_c, scale);

  // 5) output projection (WMMA, LDS-staged A panel)
  {
    dim3 grid(NTOK / 16, 8);
    out_gemm<<<grid, 256, 0, stream>>>(r_bf, wo_bf, out);
  }
}
